// InterModalityUpdate_36197984371016
// MI455X (gfx1250) — matrix operand order repaired and checked by reading, not compiled, not run on hardware
//
#include <hip/hip_runtime.h>

// ---------------------------------------------------------------------------
// InterModalityUpdate for MI455X (gfx1250): bf16 WMMA GEMMs + flash attention
// ---------------------------------------------------------------------------

typedef __attribute__((ext_vector_type(16))) __bf16 v16bf;
typedef __attribute__((ext_vector_type(8)))  __bf16 v8bf;
typedef __attribute__((ext_vector_type(8)))  float  v8f;

#define DEVINL __device__ __forceinline__

DEVINL v8f wmma_bf16(v16bf a, v16bf b, v8f c) {
  // D = A(16x32 bf16) * B(32x16 bf16) + C(16x16 f32)
  return __builtin_amdgcn_wmma_f32_16x16x32_bf16(false, a, false, b, (short)0, c,
                                                 false, false);
}

// A-fragment (16x32, bf16, row-major source [rows, ld]):
// lanes 0-15 : row=lane,    K = [k0..k0+7] and [k0+16..k0+23]
// lanes 16-31: row=lane-16, K = [k0+8..k0+15] and [k0+24..k0+31]
DEVINL v16bf ldA(const __bf16* base, int row0, int ld, int k0, int lane) {
  const __bf16* p = base + (size_t)(row0 + (lane & 15)) * ld + k0 + ((lane >> 4) << 3);
  v8bf lo = *(const v8bf*)p;
  v8bf hi = *(const v8bf*)(p + 16);
  v16bf r;
#pragma unroll
  for (int i = 0; i < 8; ++i) { r[i] = lo[i]; r[8 + i] = hi[i]; }
  return r;
}

// Same A-fragment but converting fp32 source -> bf16 on the fly.
DEVINL v16bf cvtA(const float* base, int row0, int ld, int k0, int lane) {
  const float* p = base + (size_t)(row0 + (lane & 15)) * ld + k0 + ((lane >> 4) << 3);
  v16bf r;
#pragma unroll
  for (int i = 0; i < 8; ++i) { r[i] = (__bf16)p[i]; r[8 + i] = (__bf16)p[16 + i]; }
  return r;
}

// B-fragment (32x16, bf16). Source is K-major per output column: Bt[col][k]
// lanes 0-15 : col=lane,    K = [k0..k0+7] and [k0+16..k0+23]
// lanes 16-31: col=lane-16, K = [k0+8..k0+15] and [k0+24..k0+31]
DEVINL v16bf ldB(const __bf16* base, int col0, int ld, int k0, int lane) {
  const __bf16* p = base + (size_t)(col0 + (lane & 15)) * ld + k0 + ((lane >> 4) << 3);
  v8bf lo = *(const v8bf*)p;
  v8bf hi = *(const v8bf*)(p + 16);
  v16bf r;
#pragma unroll
  for (int i = 0; i < 8; ++i) { r[i] = lo[i]; r[8 + i] = hi[i]; }
  return r;
}

DEVINL float redmax16(float x) {
#pragma unroll
  for (int m = 1; m < 16; m <<= 1) x = fmaxf(x, __shfl_xor(x, m, 16));
  return x;
}
DEVINL float redsum16(float x) {
#pragma unroll
  for (int m = 1; m < 16; m <<= 1) x += __shfl_xor(x, m, 16);
  return x;
}

// ---------------------------------------------------------------------------
// Weight transpose + fp32->bf16: W[K,N] -> Wt[N,K]   (K, N multiples of 32)
// ---------------------------------------------------------------------------
__global__ __launch_bounds__(256) void transpose_cvt(const float* __restrict__ W,
                                                     __bf16* __restrict__ Wt,
                                                     int K, int N) {
  __shared__ float t[32][33];
  int n0 = blockIdx.x * 32, k0 = blockIdx.y * 32;
  int tx = threadIdx.x, ty = threadIdx.y;
#pragma unroll
  for (int i = ty; i < 32; i += 8) t[i][tx] = W[(size_t)(k0 + i) * N + n0 + tx];
  __syncthreads();
#pragma unroll
  for (int i = ty; i < 32; i += 8)
    Wt[(size_t)(n0 + i) * K + k0 + tx] = (__bf16)t[tx][i];
}

// ---------------------------------------------------------------------------
// Head-sliced transpose of the "val" third of a trans buffer (bf16 as u16):
// T[b*L+m][4096 + h*128 + d] -> valT[(b*16+h)*128 + d][m]
// ---------------------------------------------------------------------------
__global__ __launch_bounds__(256) void transpose_heads(
    const unsigned short* __restrict__ T, unsigned short* __restrict__ valT, int L) {
  __shared__ unsigned short t[32][33];
  int bh = blockIdx.z;
  int b = bh >> 4, h = bh & 15;
  int m0 = blockIdx.x * 32, d0 = blockIdx.y * 32;
  const unsigned short* src = T + (size_t)(b * L) * 6144 + 4096 + h * 128;
  unsigned short* dst = valT + (size_t)bh * 128 * L;
  int tx = threadIdx.x, ty = threadIdx.y;
#pragma unroll
  for (int i = ty; i < 32; i += 8) t[i][tx] = src[(size_t)(m0 + i) * 6144 + d0 + tx];
  __syncthreads();
#pragma unroll
  for (int i = ty; i < 32; i += 8)
    dst[(size_t)(d0 + i) * L + m0 + tx] = t[tx][i];
}

// ---------------------------------------------------------------------------
// Fused GEMM: out = relu([A1 | A2] @ Bt^T + bias) * rowmask
//   A1: fp32 [rows, K1] (converted to bf16 on the fly), A2: bf16 [rows, K2]
//   Bt: bf16 [N, K1+K2] (pre-transposed weight), bias fp32 [N]
//   rowmask (optional) fp32 [rows]; outb (bf16) and/or outf (fp32), ld = N
// Block = 256 threads = 8 waves as 2x4; block tile 128x256; wave tile 64x64.
// K loop is split into an fp32-A phase and a bf16-A phase (no branch in loop).
// 16 WMMAs per k-step per wave; 8 fragment loads -> ~32 FLOP/byte from cache.
// ---------------------------------------------------------------------------
__global__ __launch_bounds__(256) void wmma_gemm_bias_relu(
    const float* __restrict__ A1, int K1, const __bf16* __restrict__ A2, int K2,
    const __bf16* __restrict__ Bt, const float* __restrict__ bias,
    const float* __restrict__ rowmask, __bf16* __restrict__ outb,
    float* __restrict__ outf, int N) {
  const int lane = threadIdx.x & 31;
  const int wv = threadIdx.x >> 5;
  const int wr = wv >> 2, wc = wv & 3;
  const int row0 = blockIdx.y * 128 + wr * 64;
  const int col0 = blockIdx.x * 256 + wc * 64;
  const int K = K1 + K2;

  v8f acc[4][4];
#pragma unroll
  for (int ri = 0; ri < 4; ++ri)
#pragma unroll
    for (int ci = 0; ci < 4; ++ci) acc[ri][ci] = (v8f){};

  // Phase 1: A from fp32 source (converted on the fly)
  for (int k = 0; k < K1; k += 32) {
    v16bf a[4], bb[4];
#pragma unroll
    for (int ri = 0; ri < 4; ++ri) a[ri] = cvtA(A1, row0 + 16 * ri, K1, k, lane);
#pragma unroll
    for (int ci = 0; ci < 4; ++ci) bb[ci] = ldB(Bt, col0 + 16 * ci, K, k, lane);
#pragma unroll
    for (int ri = 0; ri < 4; ++ri)
#pragma unroll
      for (int ci = 0; ci < 4; ++ci)
        acc[ri][ci] = wmma_bf16(a[ri], bb[ci], acc[ri][ci]);
  }

  // Phase 2: A from bf16 source
  for (int k = 0; k < K2; k += 32) {
    v16bf a[4], bb[4];
#pragma unroll
    for (int ri = 0; ri < 4; ++ri) a[ri] = ldA(A2, row0 + 16 * ri, K2, k, lane);
#pragma unroll
    for (int ci = 0; ci < 4; ++ci) bb[ci] = ldB(Bt, col0 + 16 * ci, K, K1 + k, lane);
#pragma unroll
    for (int ri = 0; ri < 4; ++ri)
#pragma unroll
      for (int ci = 0; ci < 4; ++ci)
        acc[ri][ci] = wmma_bf16(a[ri], bb[ci], acc[ri][ci]);
  }

  const int half = lane >> 4;
#pragma unroll
  for (int ci = 0; ci < 4; ++ci) {
    int colb = col0 + ci * 16 + (lane & 15);
    float bia = bias[colb];
#pragma unroll
    for (int ri = 0; ri < 4; ++ri) {
#pragma unroll
      for (int j = 0; j < 8; ++j) {
        int row = row0 + ri * 16 + j + half * 8;
        float x = acc[ri][ci][j] + bia;
        x = x > 0.f ? x : 0.f;
        if (rowmask) x *= rowmask[row];
        size_t idx = (size_t)row * N + colb;
        if (outb) outb[idx] = (__bf16)x;
        if (outf) outf[idx] = x;
      }
    }
  }
}

// ---------------------------------------------------------------------------
// Flash cross-attention, one wave = 16 query rows, online softmax over Lk keys.
//   Tq: queries trans buffer [B*Lq, 6144], query part at +2048 (+h*128)
//   Tk: keys    trans buffer [B*Lk, 6144], key part at +0 (+h*128)
//   valT: [B*H*128, Lk] pre-transposed values
//   mask: fp32 [B*Lk] key mask; outb: bf16 [B*Lq, 2048]
// Block = 128 threads = 4 waves; grid = (Lq/64, H, B).
// ---------------------------------------------------------------------------
__global__ __launch_bounds__(128) void flash_cross_attn(
    const __bf16* __restrict__ Tq, const __bf16* __restrict__ Tk,
    const __bf16* __restrict__ valT, const float* __restrict__ mask,
    __bf16* __restrict__ outb, int Lq, int Lk) {
  constexpr int LD = 6144, DH = 128, OS2 = 2048;
  __shared__ alignas(32) __bf16 plds[4][16][32];

  const int lane = threadIdx.x & 31;
  const int wv = threadIdx.x >> 5;
  const int b = blockIdx.z, h = blockIdx.y;
  const int qrow0 = blockIdx.x * 64 + wv * 16;
  const float scale = 0.08838834764831845f;  // 1/sqrt(128)

  const __bf16* qbase = Tq + (size_t)(b * Lq) * LD + OS2 + h * DH;
  const __bf16* kbase = Tk + (size_t)(b * Lk) * LD + h * DH;
  const __bf16* vtb = valT + (size_t)(b * 16 + h) * DH * Lk;

  v16bf qf[4];
#pragma unroll
  for (int d4 = 0; d4 < 4; ++d4) qf[d4] = ldA(qbase, qrow0, LD, d4 * 32, lane);

  v8f O[8];
#pragma unroll
  for (int t = 0; t < 8; ++t) O[t] = (v8f){};
  float mstat[8], lstat[8];
#pragma unroll
  for (int j = 0; j < 8; ++j) { mstat[j] = -1e30f; lstat[j] = 0.f; }

  const int half = lane >> 4;
  for (int key0 = 0; key0 < Lk; key0 += 32) {
    // ---- scores S[16q x 32k] = Q @ K^T ----
    v8f s0 = {}, s1 = {};
#pragma unroll
    for (int d4 = 0; d4 < 4; ++d4) {
      v16bf kf0 = ldB(kbase, key0, LD, d4 * 32, lane);
      v16bf kf1 = ldB(kbase, key0 + 16, LD, d4 * 32, lane);
      s0 = wmma_bf16(qf[d4], kf0, s0);
      s1 = wmma_bf16(qf[d4], kf1, s1);
    }
    float mk0 = mask[b * Lk + key0 + (lane & 15)];
    float mk1 = mask[b * Lk + key0 + 16 + (lane & 15)];

    // ---- online softmax (rows live across 16-lane halves) ----
    float p0[8], p1[8], corr[8];
#pragma unroll
    for (int j = 0; j < 8; ++j) {
      float a = s0[j] * scale;
      float c = s1[j] * scale;
      if (mk0 == 0.f) a = -1e30f;
      if (mk1 == 0.f) c = -1e30f;
      float cm = redmax16(fmaxf(a, c));
      float mn = fmaxf(mstat[j], cm);
      float cr = __expf(mstat[j] - mn);
      float e0 = __expf(a - mn);
      float e1 = __expf(c - mn);
      lstat[j] = lstat[j] * cr + redsum16(e0 + e1);
      mstat[j] = mn;
      corr[j] = cr;
      p0[j] = e0;
      p1[j] = e1;
    }

    // ---- reshape P (C-layout fp32) -> A-fragment (bf16) via LDS ----
    {
      int rb = half << 3, cc = lane & 15;
#pragma unroll
      for (int j = 0; j < 8; ++j) {
        plds[wv][rb + j][cc] = (__bf16)p0[j];
        plds[wv][rb + j][cc + 16] = (__bf16)p1[j];
      }
    }
    asm volatile("s_wait_dscnt 0" ::: "memory");  // per-wave LDS RAW ordering
    v16bf pf = ldA(&plds[wv][0][0], 0, 32, 0, lane);

    // ---- O = diag(corr) * O + P @ V ----
#pragma unroll
    for (int t = 0; t < 8; ++t) {
#pragma unroll
      for (int j = 0; j < 8; ++j) O[t][j] *= corr[j];
      v16bf vf = ldB(vtb, t * 16, Lk, key0, lane);
      O[t] = wmma_bf16(pf, vf, O[t]);
    }
  }

  // ---- normalize + store bf16 update ----
#pragma unroll
  for (int j = 0; j < 8; ++j) lstat[j] = lstat[j] > 0.f ? 1.f / lstat[j] : 0.f;
#pragma unroll
  for (int t = 0; t < 8; ++t) {
    int col = h * DH + t * 16 + (lane & 15);
#pragma unroll
    for (int j = 0; j < 8; ++j) {
      int rowg = b * Lq + qrow0 + j + half * 8;
      outb[(size_t)rowg * OS2 + col] = (__bf16)(O[t][j] * lstat[j]);
    }
  }
}

// ---------------------------------------------------------------------------
// Host launcher
// ---------------------------------------------------------------------------
extern "C" void kernel_launch(void* const* d_in, const int* in_sizes, int n_in,
                              void* d_out, int out_size, void* d_ws, size_t ws_size,
                              hipStream_t stream) {
  (void)in_sizes; (void)n_in; (void)out_size; (void)ws_size;

  const float* v = (const float*)d_in[0];     // [16,512,2048]
  const float* q = (const float*)d_in[1];     // [16,512,1024]
  const float* vmask = (const float*)d_in[2]; // [16,512]
  const float* qmask = (const float*)d_in[3]; // [16,512]
  const float* Wv = (const float*)d_in[4];    // [2048,6144]
  const float* bv = (const float*)d_in[5];
  const float* Wq = (const float*)d_in[6];    // [1024,6144]
  const float* bq = (const float*)d_in[7];
  const float* Wvo = (const float*)d_in[8];   // [4096,2048]
  const float* bvo = (const float*)d_in[9];
  const float* Wqo = (const float*)d_in[10];  // [3072,2048]
  const float* bqo = (const float*)d_in[11];

  float* out_v = (float*)d_out;
  float* out_q = out_v + (size_t)16 * 512 * 2048;

  char* ws = (char*)d_ws;
  size_t off = 0;
  auto take = [&](size_t bytes) {
    char* p = ws + off;
    off += (bytes + 255) & ~(size_t)255;
    return p;
  };
  __bf16* Wvt   = (__bf16*)take((size_t)6144 * 2048 * 2);
  __bf16* Wqt   = (__bf16*)take((size_t)6144 * 1024 * 2);
  __bf16* Wvot  = (__bf16*)take((size_t)2048 * 4096 * 2);
  __bf16* Wqot  = (__bf16*)take((size_t)2048 * 3072 * 2);
  __bf16* vtr   = (__bf16*)take((size_t)8192 * 6144 * 2);
  __bf16* qtr   = (__bf16*)take((size_t)8192 * 6144 * 2);
  __bf16* vvalT = (__bf16*)take((size_t)256 * 128 * 512 * 2);
  __bf16* qvalT = (__bf16*)take((size_t)256 * 128 * 512 * 2);
  __bf16* vupd  = (__bf16*)take((size_t)8192 * 2048 * 2);
  __bf16* qupd  = (__bf16*)take((size_t)8192 * 2048 * 2);
  // total workspace: 384 MiB

  dim3 tb(32, 8);
  // bf16 weight transposes (B-operand wants [N,K])
  transpose_cvt<<<dim3(6144 / 32, 2048 / 32), tb, 0, stream>>>(Wv, Wvt, 2048, 6144);
  transpose_cvt<<<dim3(6144 / 32, 1024 / 32), tb, 0, stream>>>(Wq, Wqt, 1024, 6144);
  transpose_cvt<<<dim3(2048 / 32, 4096 / 32), tb, 0, stream>>>(Wvo, Wvot, 4096, 2048);
  transpose_cvt<<<dim3(2048 / 32, 3072 / 32), tb, 0, stream>>>(Wqo, Wqot, 3072, 2048);

  // v_trans / q_trans = relu(x @ W + b) * mask, stored bf16
  // block tile 128x256
  wmma_gemm_bias_relu<<<dim3(6144 / 256, 8192 / 128), 256, 0, stream>>>(
      v, 2048, nullptr, 0, Wvt, bv, vmask, vtr, nullptr, 6144);
  wmma_gemm_bias_relu<<<dim3(6144 / 256, 8192 / 128), 256, 0, stream>>>(
      q, 1024, nullptr, 0, Wqt, bq, qmask, qtr, nullptr, 6144);

  // transpose val head slices for the P@V contraction
  transpose_heads<<<dim3(512 / 32, 128 / 32, 256), tb, 0, stream>>>(
      (const unsigned short*)vtr, (unsigned short*)vvalT, 512);
  transpose_heads<<<dim3(512 / 32, 128 / 32, 256), tb, 0, stream>>>(
      (const unsigned short*)qtr, (unsigned short*)qvalT, 512);

  // q2v: visual queries attend question keys/vals ; v2q: the reverse
  flash_cross_attn<<<dim3(512 / 64, 16, 16), 128, 0, stream>>>(
      vtr, qtr, qvalT, qmask, vupd, 512, 512);
  flash_cross_attn<<<dim3(512 / 64, 16, 16), 128, 0, stream>>>(
      qtr, vtr, vvalT, vmask, qupd, 512, 512);

  // updated_v = relu([v | v_update] @ Wvo + bvo) ; updated_q likewise
  wmma_gemm_bias_relu<<<dim3(2048 / 256, 8192 / 128), 256, 0, stream>>>(
      v, 2048, vupd, 2048, Wvot, bvo, nullptr, nullptr, out_v, 2048);
  wmma_gemm_bias_relu<<<dim3(2048 / 256, 8192 / 128), 256, 0, stream>>>(
      q, 1024, qupd, 2048, Wqot, bqo, nullptr, nullptr, out_q, 2048);
}
